// TemNonLocalBlock_26250840113384
// MI455X (gfx1250) — compile-verified, matrix operand order checked by reading, not run on hardware
//
#include <hip/hip_runtime.h>
#include <stdint.h>

typedef _Float16 v16h __attribute__((ext_vector_type(16)));
typedef float    v8f  __attribute__((ext_vector_type(8)));
typedef uint32_t v4u  __attribute__((ext_vector_type(4)));

union Frag {
    v16h     h;
    uint32_t u[8];
    v4u      q[2];
};

__device__ __forceinline__ int kmap(int g, int hi) {
    // 16-bit A/B fragment K mapping (ISA 7.12.2): lanes 0-15 vs 16-31 differ by +8
    return ((g < 4) ? (2 * g) : (16 + 2 * (g - 4))) + 8 * hi;
}

__device__ __forceinline__ uint32_t pk2(float a, float b) {
    union { _Float16 h[2]; uint32_t u; } z;
    z.h[0] = (_Float16)a; z.h[1] = (_Float16)b;
    return z.u;
}

// ---------------- prep 1: u_theta = theta_w^T cp_th ; u_phi = phi_w^T cp_ph ----
__global__ void tnl_prep_u(const float* __restrict__ th_w, const float* __restrict__ ph_w,
                           const float* __restrict__ th_b, const float* __restrict__ ph_b,
                           const float* __restrict__ cp,
                           float* __restrict__ ut, float* __restrict__ up,
                           float* __restrict__ cc) {
    int c = threadIdx.x;  // 256 threads
    float at = 0.f, ap = 0.f;
    for (int i = 0; i < 128; ++i) {
        at = fmaf(th_w[i * 256 + c], cp[i],       at);
        ap = fmaf(ph_w[i * 256 + c], cp[128 + i], ap);
    }
    ut[c] = at; up[c] = ap;
    if (c == 0) {
        float bt = 0.f, bp = 0.f;
        for (int i = 0; i < 128; ++i) {
            bt = fmaf(th_b[i], cp[i],       bt);
            bp = fmaf(ph_b[i], cp[128 + i], bp);
        }
        cc[0] = bt; cc[1] = bp;
    }
}

// ---------------- prep 2: pack g_w / W_w into WMMA A-fragment layout (f16) ----
__global__ void tnl_prep_pack(const float* __restrict__ g_w, const float* __restrict__ W_w,
                              uint32_t* __restrict__ gpk, uint32_t* __restrict__ wpk) {
    int blk = blockIdx.x;           // 0..63 g_w tiles, 64..127 W_w tiles
    int lane = threadIdx.x;         // 32 threads
    int hi = lane >> 4, lo = lane & 15;
    if (blk < 64) {
        int it = blk >> 3, kb = blk & 7;
        int row = it * 16 + lo;                       // i
        uint32_t* dst = gpk + (((size_t)(it * 8 + kb) * 32 + lane) * 8);
        #pragma unroll
        for (int g = 0; g < 8; ++g) {
            int c = kb * 32 + kmap(g, hi);
            dst[g] = pk2(g_w[row * 256 + c], g_w[row * 256 + c + 1]);
        }
    } else {
        int p = blk - 64;
        int ct = p >> 2, kb = p & 3;
        int row = ct * 16 + lo;                       // c
        uint32_t* dst = wpk + (((size_t)(ct * 4 + kb) * 32 + lane) * 8);
        #pragma unroll
        for (int g = 0; g < 8; ++g) {
            int i = kb * 32 + kmap(g, hi);
            dst[g] = pk2(W_w[row * 128 + i], W_w[row * 128 + i + 1]);
        }
    }
}

// ---------------- fused main kernel -------------------------------------------
// grid: 2048 blocks (b in [0,32) x 64 t-blocks of 16 t's), 256 threads (8 waves)
#define XSTR 258   // xs row stride (halfs), odd dword stride -> conflict-free
#define GSTR 274   // gxs row stride
#define YSTR 130   // ys row stride

__global__ void __launch_bounds__(256, 1)
tnl_main(const float* __restrict__ x, const float* __restrict__ g_b,
         const float* __restrict__ W_b, const float* __restrict__ ut,
         const float* __restrict__ up, const float* __restrict__ cc,
         const uint32_t* __restrict__ gpk, const uint32_t* __restrict__ wpk,
         float* __restrict__ out) {
    constexpr int Cc = 256, TV = 1024 * 17, MT = 272;
    extern __shared__ char smem[];
    _Float16* xs  = (_Float16*)smem;                               // [272][XSTR] (m,c)
    _Float16* gxs = (_Float16*)(smem + (size_t)MT * XSTR * 2);     // [128][GSTR] (i,m)
    float* sth = (float*)(smem + (size_t)MT * XSTR * 2 + (size_t)128 * GSTR * 2);
    float* sph = sth + MT;
    _Float16* ys = xs;                                             // alias: [272][YSTR] (m,i)

    const int tid = threadIdx.x;
    const int lane = tid & 31, wave = tid >> 5;
    const int hi = lane >> 4, lo = lane & 15;
    const int b = blockIdx.x >> 6;
    const int tb = blockIdx.x & 63;
    const long m_base = (long)tb * MT;
    const float* xb = x + (long)b * Cc * TV + m_base;    // index with c*TV + m
    float* outb = out + (long)b * Cc * TV + m_base;

    // ---- stage x tile -> LDS f16 transposed [m][c], coalesced float4 reads ----
    for (int idx = tid; idx < 256 * 68; idx += 256) {
        int c = idx / 68, j4 = idx - c * 68;
        float4 v = *(const float4*)(xb + (long)c * TV + j4 * 4);
        int m = j4 * 4;
        xs[(m + 0) * XSTR + c] = (_Float16)v.x;
        xs[(m + 1) * XSTR + c] = (_Float16)v.y;
        xs[(m + 2) * XSTR + c] = (_Float16)v.z;
        xs[(m + 3) * XSTR + c] = (_Float16)v.w;
    }
    __syncthreads();

    // ---- s_theta / s_phi : 256-dot with precomputed u vectors -----------------
    {
        const float cth = cc[0], cph = cc[1];
        for (int m = tid; m < MT; m += 256) {
            float at = cth, ap = cph;
            const _Float16* xr = xs + m * XSTR;
            #pragma unroll 4
            for (int c = 0; c < Cc; ++c) {
                float xv = (float)xr[c];
                at = fmaf(ut[c], xv, at);
                ap = fmaf(up[c], xv, ap);
            }
            sth[m] = at; sph[m] = ap;
        }
    }

    // ---- GEMM1: gx[i,m] = sum_c g_w[i,c] x[c,m]; wave w owns i-tile w ---------
    {
        const int it = wave;
        Frag aF[8];
        #pragma unroll
        for (int kb = 0; kb < 8; ++kb) {
            const v4u* p = (const v4u*)(gpk + (((size_t)(it * 8 + kb) * 32 + lane) * 8));
            aF[kb].q[0] = p[0]; aF[kb].q[1] = p[1];
        }
        for (int mt = 0; mt < 17; ++mt) {
            const int m0 = mt * 16;
            v8f acc = {};
            #pragma unroll
            for (int kb = 0; kb < 8; ++kb) {
                Frag bF;
                const _Float16* bp = xs + (m0 + lo) * XSTR + kb * 32 + 8 * hi;
                #pragma unroll
                for (int g = 0; g < 8; ++g) {
                    int k0 = (g < 4) ? 2 * g : 16 + 2 * (g - 4);
                    bF.u[g] = *(const uint32_t*)(bp + k0);
                }
                acc = __builtin_amdgcn_wmma_f32_16x16x32_f16(
                        false, aF[kb].h, false, bF.h, (short)0, acc, false, false);
            }
            #pragma unroll
            for (int r = 0; r < 8; ++r) {
                int i = it * 16 + r + 8 * hi;
                gxs[i * GSTR + m0 + lo] = (_Float16)(acc[r] + g_b[i]);
            }
        }
    }
    __syncthreads();

    // ---- per-t: f = relu(s_th (+) s_ph)/17 ; y = gx @ f^T  (K = v, padded 32) --
    {
        const float inv17 = 1.0f / 17.0f;
        for (int tt = 0; tt < 2; ++tt) {
            const int tl = wave * 2 + tt;
            const float* sT = sth + tl * 17;
            const float* sP = sph + tl * 17;
            Frag bF[2];
            #pragma unroll
            for (int nt = 0; nt < 2; ++nt) {
                int kcol = nt * 16 + lo;
                float stv = (kcol < 17) ? sT[kcol] : 0.0f;
                #pragma unroll
                for (int g = 0; g < 8; ++g) {
                    int k0 = kmap(g, hi);
                    float v0 = 0.f, v1 = 0.f;
                    if (kcol < 17) {
                        if (k0     < 17) v0 = fmaxf(stv + sP[k0],     0.0f) * inv17;
                        if (k0 + 1 < 17) v1 = fmaxf(stv + sP[k0 + 1], 0.0f) * inv17;
                    }
                    bF[nt].u[g] = pk2(v0, v1);
                }
            }
            #pragma unroll 1
            for (int it2 = 0; it2 < 8; ++it2) {
                Frag aF;
                const _Float16* gp = gxs + (it2 * 16 + lo) * GSTR + tl * 17;
                #pragma unroll
                for (int g = 0; g < 8; ++g) {
                    int k0 = kmap(g, hi);
                    union { _Float16 h[2]; uint32_t u; } z;
                    z.h[0] = (k0     < 17) ? gp[k0]     : (_Float16)0.0f;
                    z.h[1] = (k0 + 1 < 17) ? gp[k0 + 1] : (_Float16)0.0f;
                    aF.u[g] = z.u;
                }
                v8f z = {};
                v8f d0 = __builtin_amdgcn_wmma_f32_16x16x32_f16(
                            false, aF.h, false, bF[0].h, (short)0, z, false, false);
                v8f d1 = __builtin_amdgcn_wmma_f32_16x16x32_f16(
                            false, aF.h, false, bF[1].h, (short)0, z, false, false);
                #pragma unroll
                for (int r = 0; r < 8; ++r) {
                    int i = it2 * 16 + r + 8 * hi;
                    ys[(tl * 17 + lo) * YSTR + i] = (_Float16)d0[r];   // kcol = lo (<16<17)
                    if (lo == 0)
                        ys[(tl * 17 + 16) * YSTR + i] = (_Float16)d1[r]; // kcol = 16
                }
            }
        }
    }
    __syncthreads();

    // ---- GEMM2: out[c,m] = sum_i W_w[c,i] y[i,m] + W_b + x -------------------
    for (int ci = 0; ci < 2; ++ci) {
        const int ct = wave + 8 * ci;
        Frag aF[4];
        #pragma unroll
        for (int kb = 0; kb < 4; ++kb) {
            const v4u* p = (const v4u*)(wpk + (((size_t)(ct * 4 + kb) * 32 + lane) * 8));
            aF[kb].q[0] = p[0]; aF[kb].q[1] = p[1];
        }
        float wb[8];
        #pragma unroll
        for (int r = 0; r < 8; ++r) wb[r] = W_b[ct * 16 + r + 8 * hi];
        for (int mt = 0; mt < 17; ++mt) {
            const int m0 = mt * 16;
            v8f acc = {};
            #pragma unroll
            for (int kb = 0; kb < 4; ++kb) {
                Frag bF;
                const _Float16* bp = ys + (m0 + lo) * YSTR + kb * 32 + 8 * hi;
                #pragma unroll
                for (int g = 0; g < 8; ++g) {
                    int k0 = (g < 4) ? 2 * g : 16 + 2 * (g - 4);
                    bF.u[g] = *(const uint32_t*)(bp + k0);
                }
                acc = __builtin_amdgcn_wmma_f32_16x16x32_f16(
                        false, aF[kb].h, false, bF.h, (short)0, acc, false, false);
            }
            #pragma unroll
            for (int r = 0; r < 8; ++r) {
                int c = ct * 16 + r + 8 * hi;
                long off = (long)c * TV + m0 + lo;
                outb[off] = acc[r] + wb[r] + xb[off];
            }
        }
    }
}

extern "C" void kernel_launch(void* const* d_in, const int* in_sizes, int n_in,
                              void* d_out, int out_size, void* d_ws, size_t ws_size,
                              hipStream_t stream) {
    const float* x    = (const float*)d_in[0];
    const float* g_w  = (const float*)d_in[1];
    const float* g_b  = (const float*)d_in[2];
    const float* th_w = (const float*)d_in[3];
    const float* th_b = (const float*)d_in[4];
    const float* ph_w = (const float*)d_in[5];
    const float* ph_b = (const float*)d_in[6];
    const float* cp_w = (const float*)d_in[7];
    const float* W_w  = (const float*)d_in[8];
    const float* W_b  = (const float*)d_in[9];
    float* out = (float*)d_out;

    // workspace carve
    float*    ut  = (float*)d_ws;            // 256
    float*    up  = ut + 256;                // 256
    float*    cc  = up + 256;                // 2
    uint32_t* gpk = (uint32_t*)(cc + 2);     // 8*8*32*8   = 16384 dwords
    uint32_t* wpk = gpk + 16384;             // 16*4*32*8  = 16384 dwords

    tnl_prep_u<<<1, 256, 0, stream>>>(th_w, ph_w, th_b, ph_b, cp_w, ut, up, cc);
    tnl_prep_pack<<<128, 32, 0, stream>>>(g_w, W_w, gpk, wpk);

    size_t lds = (size_t)272 * XSTR * 2 + (size_t)128 * GSTR * 2 + 2 * 272 * sizeof(float);
    tnl_main<<<32 * 64, 256, lds, stream>>>(x, g_b, W_b, ut, up, cc, gpk, wpk, out);
}
